// TGCN_56057913147771
// MI455X (gfx1250) — matrix operand order repaired and checked by reading, not compiled, not run on hardware
//
#include <hip/hip_runtime.h>
#include <hip/hip_bf16.h>
#include <math.h>

typedef __attribute__((ext_vector_type(16))) _Float16 v16h;
typedef __attribute__((ext_vector_type(8)))  _Float16 v8h;
typedef __attribute__((ext_vector_type(8)))  float    v8f;

// Problem constants (match reference)
static constexpr int  Bb   = 32;
static constexpr int  Tt   = 12;
static constexpr int  Nn   = 5000;
static constexpr int  INF_ = 2;
static constexpr int  Hh   = 128;
static constexpr int  HORZ = 12;
static constexpr int  Ee   = 80000;
static constexpr int  NT   = Bb * Nn;      // 160000
static constexpr int  LDSROW = 136;        // 128 halfs + 8 pad -> 272B row stride, bank-friendly
static constexpr int  F32ROW = 132;        // 128 floats + 4 pad -> 528B row stride
static constexpr int  MATSZ  = 4 * 8 * 32 * 16; // 16384 halfs per packed 128x128 weight
// Graph edges index nodes [0, Nn) only -> propagation buffers are nonzero only in the
// first Nn rows. Round up to a whole 16-row tile boundary.
static constexpr int  NACT       = 5008;       // 313 * 16
static constexpr int  ACT_TILES  = NACT / 16;  // 313

__device__ __forceinline__ v8f wmma16(v16h a, v16h b, v8f c) {
    return __builtin_amdgcn_wmma_f32_16x16x32_f16(false, a, false, b, (short)0, c, false, false);
}

// Async (ASYNCcnt-tracked) 16B memory->LDS copy, gfx1250 path.
__device__ __forceinline__ void async_ld16(unsigned lds_off, const void* gaddr) {
    asm volatile("global_load_async_to_lds_b128 %0, %1, off"
                 :: "v"(lds_off), "v"((unsigned long long)gaddr) : "memory");
}
__device__ __forceinline__ void wait_async0() {
    asm volatile("s_wait_asynccnt 0x0" ::: "memory");
}

// ---------------- utility kernels ----------------

__global__ __launch_bounds__(256) void zero_f4(float4* p, int n4) {
    int i = blockIdx.x * blockDim.x + threadIdx.x;
    if (i < n4) p[i] = make_float4(0.f, 0.f, 0.f, 0.f);
}

__global__ __launch_bounds__(256) void deg_scatter(const int* __restrict__ ei, float* __restrict__ deg) {
    int e = blockIdx.x * blockDim.x + threadIdx.x;
    if (e < Ee) atomicAdd(&deg[ei[e]], 1.0f);
}

__global__ __launch_bounds__(256) void deg_to_dis(float* __restrict__ d, int n) {
    int i = blockIdx.x * blockDim.x + threadIdx.x;
    if (i < n) {
        float v = d[i];
        d[i] = (v > 0.f) ? rsqrtf(fmaxf(v, 1.f)) : 0.f;
    }
}

__global__ __launch_bounds__(256) void norm_kernel(const int* __restrict__ ei, const float* __restrict__ dis,
                                                   float* __restrict__ nrm) {
    int e = blockIdx.x * blockDim.x + threadIdx.x;
    if (e < Ee) nrm[e] = -dis[ei[e]] * dis[ei[e + Ee]];
}

// ---------------- CSR build (group edges by dst, one time) ----------------

__global__ __launch_bounds__(256) void csr_count(const int* __restrict__ ei, int* __restrict__ cnt) {
    int e = blockIdx.x * blockDim.x + threadIdx.x;
    if (e < Ee) atomicAdd(&cnt[ei[e + Ee]], 1);
}

// Single-block exclusive scan over Nn counts -> base[0..Nn]
__global__ __launch_bounds__(1024) void csr_scan(const int* __restrict__ cnt, int* __restrict__ base) {
    __shared__ int psum[1024];
    int tid = threadIdx.x;
    const int CH = (Nn + 1023) / 1024;   // 5
    int start = tid * CH;
    int end   = (start + CH < Nn) ? start + CH : Nn;
    int s = 0;
    for (int i = start; i < end; i++) s += cnt[i];
    psum[tid] = s;
    __syncthreads();
    for (int o = 1; o < 1024; o <<= 1) {
        int v = (tid >= o) ? psum[tid - o] : 0;
        __syncthreads();
        psum[tid] += v;
        __syncthreads();
    }
    int excl = (tid == 0) ? 0 : psum[tid - 1];
    for (int i = start; i < end; i++) {
        base[i] = excl;
        excl += cnt[i];
    }
    if (tid == 1023) base[Nn] = psum[1023];
}

__global__ __launch_bounds__(256) void csr_fill(const int* __restrict__ ei, const float* __restrict__ nrm,
                                                const int* __restrict__ base, int* __restrict__ cursor,
                                                int* __restrict__ csr_src, float* __restrict__ csr_nrm) {
    int e = blockIdx.x * blockDim.x + threadIdx.x;
    if (e >= Ee) return;
    int d = ei[e + Ee];
    int pos = atomicAdd(&cursor[d], 1);
    int idx = base[d] + pos;
    csr_src[idx] = ei[e];
    csr_nrm[idx] = nrm[e];
}

// Gather timestep slice x_t: [NT,2] from x[B,T,N,IN]
__global__ __launch_bounds__(256) void gather_xt(const float* __restrict__ x, float* __restrict__ xt, int t) {
    int row = blockIdx.x * blockDim.x + threadIdx.x;
    if (row >= NT) return;
    int b = row / Nn, n = row % Nn;
    const float* p = x + (((size_t)b * Tt + t) * Nn + n) * INF_;
    xt[row * 2 + 0] = p[0];
    xt[row * 2 + 1] = p[1];
}

// 2-feature CSR gather: out[d] = sum_j norm_j * in[src_j]   (one thread per dst node)
__global__ __launch_bounds__(256) void prop_x_gather(const int* __restrict__ base, const int* __restrict__ csr_src,
                                                     const float* __restrict__ csr_nrm,
                                                     const float* __restrict__ tin, float* __restrict__ tout) {
    int node = blockIdx.x * blockDim.x + threadIdx.x;
    if (node >= Nn) return;
    int jb = base[node], je = base[node + 1];
    float a0 = 0.f, a1 = 0.f;
    for (int j = jb; j < je; j++) {
        int s = csr_src[j];
        float nm = csr_nrm[j];
        a0 += nm * tin[s * 2 + 0];
        a1 += nm * tin[s * 2 + 1];
    }
    tout[node * 2 + 0] = a0;
    tout[node * 2 + 1] = a1;
}

// 128-feature CSR gather: one wave per dst node, 4 feats/lane, no atomics, no zero pass
__global__ __launch_bounds__(256) void prop_h_gather(const int* __restrict__ base, const int* __restrict__ csr_src,
                                                     const float* __restrict__ csr_nrm,
                                                     const float* __restrict__ tin, float* __restrict__ tout) {
    int gid = blockIdx.x * blockDim.x + threadIdx.x;
    int node = gid >> 5;
    int l = gid & 31;
    if (node >= Nn) return;
    int jb = base[node], je = base[node + 1];
    float4 acc = make_float4(0.f, 0.f, 0.f, 0.f);
    for (int j = jb; j < je; j++) {
        int s = csr_src[j];
        float nm = csr_nrm[j];
        float4 v = *(const float4*)(tin + (size_t)s * Hh + l * 4);
        acc.x += nm * v.x; acc.y += nm * v.y; acc.z += nm * v.z; acc.w += nm * v.w;
    }
    *(float4*)(tout + (size_t)node * Hh + l * 4) = acc;
}

// Pack a 128x128 weight into WMMA B-fragment layout:
// packed[mat][kt][nt][lane][hh] ; k = kt*32 + (lane<16?0:16) + hh ; n = nt*16 + (lane&15)
__global__ __launch_bounds__(256) void pack_weights(const float* __restrict__ W2, const float* __restrict__ Wz,
                                                    const float* __restrict__ Wr, const float* __restrict__ Wc,
                                                    _Float16* __restrict__ packed) {
    int idx = blockIdx.x * blockDim.x + threadIdx.x;
    if (idx >= 9 * MATSZ) return;
    int mat  = idx / MATSZ;
    int rem  = idx % MATSZ;
    int hh   = rem & 15;
    int lane = (rem >> 4) & 31;
    int nt   = (rem >> 9) & 7;
    int kt   = (rem >> 12) & 3;
    int k = kt * 32 + ((lane < 16) ? 0 : 16) + hh;
    int n = nt * 16 + (lane & 15);
    float v = 0.f;
    switch (mat) {
        case 0: v = W2[k * Hh + n] - W2[(2 * Hh + k) * Hh + n]; break; // W2[0]-W2[2]
        case 1: v = W2[(Hh + k) * Hh + n];                      break; // W2[1]
        case 2: v = 2.f * W2[(2 * Hh + k) * Hh + n];            break; // 2*W2[2]
        case 3: v = Wz[k * Hh + n];                             break;
        case 4: v = Wz[(Hh + k) * Hh + n];                      break;
        case 5: v = Wr[k * Hh + n];                             break;
        case 6: v = Wr[(Hh + k) * Hh + n];                      break;
        case 7: v = Wc[k * Hh + n];                             break;
        case 8: v = Wc[(Hh + k) * Hh + n];                      break;
    }
    packed[idx] = (_Float16)v;
}

// ---------------- WMMA GEMM helpers ----------------

__device__ __forceinline__ void stage_cvt_f32(_Float16 (*As)[LDSROW], const float* __restrict__ src,
                                              int row0, int tid) {
    int r = tid >> 4;
    int c = (tid & 15) * 8;
    const float4* p = (const float4*)(src + (size_t)(row0 + r) * Hh + c);
    float4 a = p[0], b = p[1];
    _Float16* d = &As[r][c];
    d[0] = (_Float16)a.x; d[1] = (_Float16)a.y; d[2] = (_Float16)a.z; d[3] = (_Float16)a.w;
    d[4] = (_Float16)b.x; d[5] = (_Float16)b.y; d[6] = (_Float16)b.z; d[7] = (_Float16)b.w;
}

// A fragment (16x32 f16) for K-tile kt from LDS-staged row-major tile
__device__ __forceinline__ v16h load_afrag(const _Float16 (*As)[LDSROW], int kt, int lane) {
    int m = lane & 15;
    int base = kt * 32 + ((lane < 16) ? 0 : 16);
    v8h lo = *(const v8h*)&As[m][base];
    v8h hi = *(const v8h*)&As[m][base + 8];
    return __builtin_shufflevector(lo, hi, 0, 1, 2, 3, 4, 5, 6, 7, 8, 9, 10, 11, 12, 13, 14, 15);
}

__device__ __forceinline__ v16h load_bfrag(const _Float16* __restrict__ pm, int kt, int nt, int lane) {
    return *(const v16h*)(pm + (((kt * 8 + nt) * 32 + lane) << 4));
}

// ---------------- fully fused GRU cell (one 16-row x 128-col strip per block) ----------------
//  Phase 0: stage h (f16+f32), P1,P2 (active only), async-stage xt/xp row slices; ic tile in-LDS
//  Phase 1: hc = h@(W0-W2) [+ P1@W1 + P2@(2W2)] + b2     (WMMA, fp32 acc kept in regs)
//  Phase 2: z = sig(ic@WzT + hc@WzB + bz), r = sig(... Wr ...); rh = r*hc -> LDS
//  Phase 3: h~ = tanh(ic@WcT + rh@WcB + bc); h = z*h + (1-z)*h~   (in place, own rows only)
__global__ __launch_bounds__(256) void fused_step(const float* __restrict__ P1, const float* __restrict__ P2,
                                                  const float* __restrict__ xt, const float* __restrict__ xp1,
                                                  const float* __restrict__ xp2,
                                                  const float* __restrict__ W1, const float* __restrict__ b1,
                                                  const _Float16* __restrict__ packed,
                                                  const float* __restrict__ b2, const float* __restrict__ bz,
                                                  const float* __restrict__ br, const float* __restrict__ bc,
                                                  float* __restrict__ h) {
    __shared__ __align__(32) _Float16 AsH[16][LDSROW];   // h (f16)
    __shared__ __align__(32) _Float16 AsP1[16][LDSROW];  // P1 -> later hc (f16)
    __shared__ __align__(32) _Float16 AsP2[16][LDSROW];  // P2 -> later rh (f16)
    __shared__ __align__(32) _Float16 AsIC[16][LDSROW];  // ic (f16)
    __shared__ __align__(16) float    Hs[16][F32ROW];    // h (f32) for exact update
    __shared__ __align__(16) float    Xs[3][32];         // xt/xp1/xp2 row slices (16 rows x 2)

    int tid = threadIdx.x;
    int row0 = blockIdx.x * 16;
    bool active = (blockIdx.x < ACT_TILES);

    // Async f32 row-slice staging (no conversion needed -> async path)
    if (tid < 8) {
        async_ld16((unsigned)(unsigned long long)&Xs[0][tid * 4], xt + (size_t)row0 * 2 + tid * 4);
    } else if (tid < 16 && active) {
        async_ld16((unsigned)(unsigned long long)&Xs[1][(tid - 8) * 4], xp1 + (size_t)row0 * 2 + (tid - 8) * 4);
    } else if (tid < 24 && active) {
        async_ld16((unsigned)(unsigned long long)&Xs[2][(tid - 16) * 4], xp2 + (size_t)row0 * 2 + (tid - 16) * 4);
    }

    // Stage h: f16 for WMMA + f32 copy for the state update
    {
        int r = tid >> 4;
        int c = (tid & 15) * 8;
        const float4* p = (const float4*)(h + (size_t)(row0 + r) * Hh + c);
        float4 a = p[0], b = p[1];
        *(float4*)&Hs[r][c]     = a;
        *(float4*)&Hs[r][c + 4] = b;
        _Float16* d = &AsH[r][c];
        d[0] = (_Float16)a.x; d[1] = (_Float16)a.y; d[2] = (_Float16)a.z; d[3] = (_Float16)a.w;
        d[4] = (_Float16)b.x; d[5] = (_Float16)b.y; d[6] = (_Float16)b.z; d[7] = (_Float16)b.w;
    }
    if (active) {
        stage_cvt_f32(AsP1, P1, row0, tid);
        stage_cvt_f32(AsP2, P2, row0, tid);
    }
    wait_async0();
    __syncthreads();

    // ic tile in-LDS: ic = x@(W1[0]-W1[2]) + xp1@W1[1] + 2*xp2@W1[2] + b1   (K=2)
    {
        int r  = tid >> 4;
        int c0 = (tid & 15) * 8;
        float x0 = Xs[0][r * 2], x1 = Xs[0][r * 2 + 1];
        float p10 = 0.f, p11 = 0.f, p20 = 0.f, p21 = 0.f;
        if (active) {
            p10 = Xs[1][r * 2]; p11 = Xs[1][r * 2 + 1];
            p20 = Xs[2][r * 2]; p21 = Xs[2][r * 2 + 1];
        }
#pragma unroll
        for (int j = 0; j < 8; j++) {
            int col = c0 + j;
            float w00 = W1[0 * Hh + col], w01 = W1[1 * Hh + col];
            float w10 = W1[2 * Hh + col], w11 = W1[3 * Hh + col];
            float w20 = W1[4 * Hh + col], w21 = W1[5 * Hh + col];
            float acc = b1[col] + x0 * (w00 - w20) + x1 * (w01 - w21)
                      + p10 * w10 + p11 * w11 + 2.f * (p20 * w20 + p21 * w21);
            AsIC[r][col] = (_Float16)acc;
        }
    }

    int w = tid >> 5, lane = tid & 31;
    int n = lane & 15;
    int col = w * 16 + n;
    int mb = (lane < 16) ? 0 : 8;

    // Phase 1: hc
    v8f chc = {};
#pragma unroll
    for (int kt = 0; kt < 4; kt++) {
        chc = wmma16(load_afrag(AsH, kt, lane), load_bfrag(packed, kt, w, lane), chc);
    }
    if (active) {
#pragma unroll
        for (int kt = 0; kt < 4; kt++) {
            chc = wmma16(load_afrag(AsP1, kt, lane), load_bfrag(packed + 1 * MATSZ, kt, w, lane), chc);
        }
#pragma unroll
        for (int kt = 0; kt < 4; kt++) {
            chc = wmma16(load_afrag(AsP2, kt, lane), load_bfrag(packed + 2 * MATSZ, kt, w, lane), chc);
        }
    }
    float hcv[8];
    {
        float b2v = b2[col];
#pragma unroll
        for (int i = 0; i < 8; i++) hcv[i] = chc[i] + b2v;
    }
    __syncthreads();                        // all P1/P2/ic-tile accesses of phase 1 done
#pragma unroll
    for (int i = 0; i < 8; i++) AsP1[mb + i][col] = (_Float16)hcv[i];   // hc tile over P1 slot
    __syncthreads();

    // Phase 2: gates (shared A fragments for z and r)
    v8f cz = {}, cr = {};
#pragma unroll
    for (int kt = 0; kt < 4; kt++) {
        v16h a = load_afrag(AsIC, kt, lane);
        cz = wmma16(a, load_bfrag(packed + 3 * MATSZ, kt, w, lane), cz);
        cr = wmma16(a, load_bfrag(packed + 5 * MATSZ, kt, w, lane), cr);
    }
#pragma unroll
    for (int kt = 0; kt < 4; kt++) {
        v16h a = load_afrag(AsP1, kt, lane);
        cz = wmma16(a, load_bfrag(packed + 4 * MATSZ, kt, w, lane), cz);
        cr = wmma16(a, load_bfrag(packed + 6 * MATSZ, kt, w, lane), cr);
    }
    float zv[8];
    {
        float bzv = bz[col], brv = br[col];
#pragma unroll
        for (int i = 0; i < 8; i++) {
            zv[i] = 1.f / (1.f + __expf(-(cz[i] + bzv)));
            float rv = 1.f / (1.f + __expf(-(cr[i] + brv)));
            AsP2[mb + i][col] = (_Float16)(rv * hcv[i]);   // rh tile over P2 slot
        }
    }
    __syncthreads();

    // Phase 3: candidate + state update
    v8f cc = {};
#pragma unroll
    for (int kt = 0; kt < 4; kt++) {
        cc = wmma16(load_afrag(AsIC, kt, lane), load_bfrag(packed + 7 * MATSZ, kt, w, lane), cc);
    }
#pragma unroll
    for (int kt = 0; kt < 4; kt++) {
        cc = wmma16(load_afrag(AsP2, kt, lane), load_bfrag(packed + 8 * MATSZ, kt, w, lane), cc);
    }
    {
        float bcv = bc[col];
#pragma unroll
        for (int i = 0; i < 8; i++) {
            float ht   = tanhf(cc[i] + bcv);
            float hold = Hs[mb + i][col];
            h[(size_t)(row0 + mb + i) * Hh + col] = zv[i] * hold + (1.f - zv[i]) * ht;
        }
    }
}

// out[b,hor,n] = h[b*N+n,:] @ Wo[:,hor] + bo[hor]
__global__ __launch_bounds__(256) void out_kernel(const float* __restrict__ h, const float* __restrict__ Wo,
                                                  const float* __restrict__ bo, float* __restrict__ out) {
    int row = blockIdx.x * blockDim.x + threadIdx.x;
    if (row >= NT) return;
    float acc[HORZ];
#pragma unroll
    for (int j = 0; j < HORZ; j++) acc[j] = bo[j];
    for (int k = 0; k < Hh; k += 4) {
        float4 hv = *(const float4*)(h + (size_t)row * Hh + k);
#pragma unroll
        for (int j = 0; j < HORZ; j++) {
            acc[j] += hv.x * Wo[(k + 0) * HORZ + j] + hv.y * Wo[(k + 1) * HORZ + j] +
                      hv.z * Wo[(k + 2) * HORZ + j] + hv.w * Wo[(k + 3) * HORZ + j];
        }
    }
    int b = row / Nn, n = row % Nn;
#pragma unroll
    for (int j = 0; j < HORZ; j++) out[((size_t)b * HORZ + j) * Nn + n] = acc[j];
}

// ---------------- launcher ----------------

extern "C" void kernel_launch(void* const* d_in, const int* in_sizes, int n_in,
                              void* d_out, int out_size, void* d_ws, size_t ws_size,
                              hipStream_t stream) {
    const float* x  = (const float*)d_in[0];
    const int*   ei = (const int*)d_in[1];
    const float* W1 = (const float*)d_in[2];
    const float* b1 = (const float*)d_in[3];
    const float* W2 = (const float*)d_in[4];
    const float* b2 = (const float*)d_in[5];
    const float* Wz = (const float*)d_in[6];
    const float* bz = (const float*)d_in[7];
    const float* Wr = (const float*)d_in[8];
    const float* br = (const float*)d_in[9];
    const float* Wc = (const float*)d_in[10];
    const float* bc = (const float*)d_in[11];
    const float* Wo = (const float*)d_in[12];
    const float* bo = (const float*)d_in[13];
    float* out = (float*)d_out;

    char* wsp = (char*)d_ws;
    size_t off = 0;
    auto carve = [&](size_t bytes) -> void* {
        void* p = wsp + off;
        off = (off + bytes + 255) & ~(size_t)255;
        return p;
    };
    float*     nrm     = (float*)carve((size_t)Ee * 4);
    float*     dis     = (float*)carve((size_t)NT * 4);
    float*     xt      = (float*)carve((size_t)NT * 2 * 4);
    float*     xp1     = (float*)carve((size_t)NT * 2 * 4);
    float*     xp2     = (float*)carve((size_t)NT * 2 * 4);
    float*     h       = (float*)carve((size_t)NT * Hh * 4);
    float*     P1      = (float*)carve((size_t)NT * Hh * 4);
    float*     P2      = (float*)carve((size_t)NT * Hh * 4);
    _Float16*  packed  = (_Float16*)carve((size_t)9 * MATSZ * 2);
    int*       cnt     = (int*)carve((size_t)Nn * 4);
    int*       base    = (int*)carve((size_t)(Nn + 1) * 4);
    int*       cursor  = (int*)carve((size_t)Nn * 4);
    int*       csr_src = (int*)carve((size_t)Ee * 4);
    float*     csr_nrm = (float*)carve((size_t)Ee * 4);
    (void)ws_size; (void)in_sizes; (void)n_in; (void)out_size;

    const int NB = 256;
    const int nRowTiles = NT / 16;              // 10000
    const int nH4 = NT * Hh / 4;                // full [NT,H] buffer in float4

    // --- One-time setup ---
    // Normalization
    zero_f4<<<(NT / 4 + NB - 1) / NB, NB, 0, stream>>>((float4*)dis, NT / 4);
    deg_scatter<<<(Ee + NB - 1) / NB, NB, 0, stream>>>(ei, dis);
    deg_to_dis<<<(NT + NB - 1) / NB, NB, 0, stream>>>(dis, NT);
    norm_kernel<<<(Ee + NB - 1) / NB, NB, 0, stream>>>(ei, dis, nrm);
    // CSR (edges grouped by dst) for gather-style propagation
    zero_f4<<<(Nn / 4 + NB - 1) / NB, NB, 0, stream>>>((float4*)cnt, Nn / 4);
    csr_count<<<(Ee + NB - 1) / NB, NB, 0, stream>>>(ei, cnt);
    csr_scan<<<1, 1024, 0, stream>>>(cnt, base);
    zero_f4<<<(Nn / 4 + NB - 1) / NB, NB, 0, stream>>>((float4*)cursor, Nn / 4);
    csr_fill<<<(Ee + NB - 1) / NB, NB, 0, stream>>>(ei, nrm, base, cursor, csr_src, csr_nrm);
    // Weights + initial state
    pack_weights<<<(9 * MATSZ + NB - 1) / NB, NB, 0, stream>>>(W2, Wz, Wr, Wc, packed);
    zero_f4<<<(nH4 + NB - 1) / NB, NB, 0, stream>>>((float4*)h, nH4);
    // Rows [Nn, NACT) of the propagation buffers are read by active tiles but never
    // written by the gathers -> zero them once.
    zero_f4<<<1, NB, 0, stream>>>((float4*)(P1 + (size_t)Nn * Hh), (NACT - Nn) * Hh / 4);
    zero_f4<<<1, NB, 0, stream>>>((float4*)(P2 + (size_t)Nn * Hh), (NACT - Nn) * Hh / 4);
    zero_f4<<<1, NB, 0, stream>>>((float4*)(xp1 + (size_t)Nn * 2), (NACT - Nn) * 2 / 4);
    zero_f4<<<1, NB, 0, stream>>>((float4*)(xp2 + (size_t)Nn * 2), (NACT - Nn) * 2 / 4);

    // --- Recurrence ---
    for (int t = 0; t < Tt; t++) {
        gather_xt<<<(NT + NB - 1) / NB, NB, 0, stream>>>(x, xt, t);

        // Chebyshev propagation on the 2-feature input (CSR gather, no atomics)
        prop_x_gather<<<(Nn + NB - 1) / NB, NB, 0, stream>>>(base, csr_src, csr_nrm, xt, xp1);
        prop_x_gather<<<(Nn + NB - 1) / NB, NB, 0, stream>>>(base, csr_src, csr_nrm, xp1, xp2);

        // Chebyshev propagation on the (pre-update) hidden state (CSR gather, one wave/node)
        prop_h_gather<<<(Nn * 32 + NB - 1) / NB, NB, 0, stream>>>(base, csr_src, csr_nrm, h, P1);
        prop_h_gather<<<(Nn * 32 + NB - 1) / NB, NB, 0, stream>>>(base, csr_src, csr_nrm, P1, P2);

        // One fused WMMA kernel: ic tile + hc + gates + candidate + GRU state update
        fused_step<<<nRowTiles, NB, 0, stream>>>(P1, P2, xt, xp1, xp2, W1, b1,
                                                 packed, b2, bz, br, bc, h);
    }

    out_kernel<<<(NT + NB - 1) / NB, NB, 0, stream>>>(h, Wo, bo, out);
}
